// IthemalRNN_76424648065731
// MI455X (gfx1250) — matrix operand order repaired and checked by reading, not compiled, not run on hardware
//
#include <hip/hip_runtime.h>
#include <hip/hip_bf16.h>
#include <cstdint>
#include <cstddef>

// Problem constants (from the reference): V=2048, E=H=256, B=256, T=16, MMAX=96
#define TMAX 16
#define EDIM 256
#define HDIM 256
#define GDIM 1024   // 4*H
#define MMAX 96
#define VDIM 2048
#define HPAD 264    // padded bf16 row stride for LDS h buffer (bank spread)
#define CPAD 260    // padded f32 row stride for LDS c buffer
#define MROWS 32    // sequences per block (2 row-tiles of 16)

typedef __attribute__((ext_vector_type(16))) __bf16 v16bf;
typedef __attribute__((ext_vector_type(8)))  float  v8f;

union FragBF {
  v16bf v;
  unsigned short u[16];
  uint4 q[2];
};

__device__ __forceinline__ unsigned short f2bf(float f) {
  union { float f; uint32_t u; } x; x.f = f;
  uint32_t r = x.u + 0x7FFFu + ((x.u >> 16) & 1u);   // round-to-nearest-even
  return (unsigned short)(r >> 16);
}
__device__ __forceinline__ float bf2f(unsigned short b) {
  union { uint32_t u; float f; } x; x.u = ((uint32_t)b) << 16;
  return x.f;
}
__device__ __forceinline__ float sigmoidf_(float x) {
  return 1.0f / (1.0f + __expf(-x));
}

// ---------------------------------------------------------------- prep kernels
__global__ void cvt_f32_bf16(const float* __restrict__ src,
                             unsigned short* __restrict__ dst, int n) {
  int i = blockIdx.x * blockDim.x + threadIdx.x;
  if (i < n) dst[i] = f2bf(src[i]);
}

__global__ void add_bias(const float* __restrict__ a, const float* __restrict__ b,
                         float* __restrict__ o, int n) {
  int i = blockIdx.x * blockDim.x + threadIdx.x;
  if (i < n) o[i] = a[i] + b[i];
}

// --------------------------------------------------------------- token LSTM
// One block = 32 sequences (two 16-row WMMA tiles).  8 waves; wave w owns gate
// columns { g*256 + w*32 + tc*16 + lane }.  Each B fragment load feeds TWO
// wmmas (row-tile 0 and 1).  Cell state c lives in LDS to cap VGPR pressure.
__global__ __launch_bounds__(256)
void token_lstm_kernel(const int* __restrict__ tokens,         // [N,T]
                       const int* __restrict__ tok_len,        // [N]
                       const unsigned short* __restrict__ embbf, // [V,E] bf16
                       const unsigned short* __restrict__ wih, // [4H,E] bf16
                       const unsigned short* __restrict__ whh, // [4H,H] bf16
                       const float* __restrict__ bias,         // [4H]
                       unsigned short* __restrict__ ihid,      // [N,H] bf16
                       int N) {
  __shared__ unsigned short hbuf[MROWS][HPAD];  // h_{t-1}, bf16
  __shared__ float cbuf[MROWS][CPAD];           // cell state, f32
  __shared__ int len_s[MROWS];

  const int tid  = threadIdx.x;
  const int lane = tid & 31;
  const int wave = tid >> 5;
  const int l    = lane & 15;
  const int half = lane >> 4;
  const int seqbase = blockIdx.x * MROWS;

  float breg[8];
#pragma unroll
  for (int j = 0; j < 8; ++j) {
    const int gi = j >> 1, tc = j & 1;
    breg[j] = bias[gi * HDIM + wave * 32 + tc * 16 + l];
  }

  for (int e = tid; e < MROWS * HPAD; e += 256) ((unsigned short*)hbuf)[e] = 0;
  for (int e = tid; e < MROWS * CPAD; e += 256) ((float*)cbuf)[e] = 0.0f;
  if (tid < MROWS) {
    int s = seqbase + tid; if (s > N - 1) s = N - 1;
    len_s[tid] = tok_len[s];
  }
  __syncthreads();

  int row0 = seqbase + l;      if (row0 > N - 1) row0 = N - 1;
  int row1 = seqbase + 16 + l; if (row1 > N - 1) row1 = N - 1;

  // per-lane weight base pointers; the 8 column tiles live at immediate offsets
  const unsigned short* wb_ih = wih + (size_t)(wave * 32 + l) * EDIM;
  const unsigned short* wb_hh = whh + (size_t)(wave * 32 + l) * HDIM;

  for (int t = 0; t < TMAX; ++t) {
    v8f acc[2][8];
#pragma unroll
    for (int mt = 0; mt < 2; ++mt)
#pragma unroll
      for (int j = 0; j < 8; ++j)
#pragma unroll
        for (int r = 0; r < 8; ++r) acc[mt][j][r] = 0.0f;

    // -------- input path: x_t @ W_ih^T (x rows = bf16 embedding rows) -------
    const unsigned short* er0 = embbf + (size_t)tokens[row0 * TMAX + t] * EDIM;
    const unsigned short* er1 = embbf + (size_t)tokens[row1 * TMAX + t] * EDIM;
    __builtin_prefetch(er0, 0, 0);
    __builtin_prefetch(er1, 0, 0);

#pragma unroll 1
    for (int kk = 0; kk < EDIM / 32; ++kk) {
      const int base1 = kk * 32 + half * 8;     // A 16x32 bf16 lane layout
      const int base2 = base1 + 16;
      const int wo    = kk * 32 + half * 8;     // B 32x16: same K split as A
      FragBF a0, a1;
      a0.q[0] = *(const uint4*)(er0 + base1);
      a0.q[1] = *(const uint4*)(er0 + base2);
      a1.q[0] = *(const uint4*)(er1 + base1);
      a1.q[1] = *(const uint4*)(er1 + base2);
#pragma unroll
      for (int j = 0; j < 8; ++j) {
        const int gi = j >> 1, tc = j & 1;
        const unsigned short* wr = wb_ih + (size_t)(gi * 256 + tc * 16) * EDIM + wo;
        FragBF bf;
        bf.q[0] = *(const uint4*)wr;
        bf.q[1] = *(const uint4*)(wr + 16);
        acc[0][j] = __builtin_amdgcn_wmma_f32_16x16x32_bf16(
            false, a0.v, false, bf.v, (short)0, acc[0][j], false, false);
        acc[1][j] = __builtin_amdgcn_wmma_f32_16x16x32_bf16(
            false, a1.v, false, bf.v, (short)0, acc[1][j], false, false);
      }
    }

    // -------- recurrent path: h_{t-1} @ W_hh^T (A from LDS) --------
#pragma unroll 1
    for (int kk = 0; kk < HDIM / 32; ++kk) {
      const int base1 = kk * 32 + half * 8;
      const int base2 = base1 + 16;
      const int wo    = kk * 32 + half * 8;
      FragBF a0, a1;
      a0.q[0] = *(const uint4*)&hbuf[l][base1];
      a0.q[1] = *(const uint4*)&hbuf[l][base2];
      a1.q[0] = *(const uint4*)&hbuf[16 + l][base1];
      a1.q[1] = *(const uint4*)&hbuf[16 + l][base2];
#pragma unroll
      for (int j = 0; j < 8; ++j) {
        const int gi = j >> 1, tc = j & 1;
        const unsigned short* wr = wb_hh + (size_t)(gi * 256 + tc * 16) * HDIM + wo;
        FragBF bf;
        bf.q[0] = *(const uint4*)wr;
        bf.q[1] = *(const uint4*)(wr + 16);
        acc[0][j] = __builtin_amdgcn_wmma_f32_16x16x32_bf16(
            false, a0.v, false, bf.v, (short)0, acc[0][j], false, false);
        acc[1][j] = __builtin_amdgcn_wmma_f32_16x16x32_bf16(
            false, a1.v, false, bf.v, (short)0, acc[1][j], false, false);
      }
    }

    __syncthreads();   // all waves done reading hbuf

    // -------- LSTM cell update; this lane owns 32 (row,hh) cells --------
#pragma unroll
    for (int mt = 0; mt < 2; ++mt) {
#pragma unroll
      for (int tc = 0; tc < 2; ++tc) {
        const int hh = wave * 32 + tc * 16 + l;
#pragma unroll
        for (int r = 0; r < 8; ++r) {
          const int row = mt * 16 + r + 8 * half;   // C/D layout: M = r + 8*half
          const float ig = acc[mt][0 + tc][r] + breg[0 + tc];
          const float fg = acc[mt][2 + tc][r] + breg[2 + tc];
          const float gg = acc[mt][4 + tc][r] + breg[4 + tc];
          const float og = acc[mt][6 + tc][r] + breg[6 + tc];
          const float co = cbuf[row][hh];
          const float cn = sigmoidf_(fg) * co + sigmoidf_(ig) * tanhf(gg);
          const float hn = sigmoidf_(og) * tanhf(cn);
          if (t < len_s[row]) {                     // sequence-length mask
            cbuf[row][hh] = cn;
            hbuf[row][hh] = f2bf(hn);
          }
        }
      }
    }
    __syncthreads();
  }

  // final hidden states become the instruction-LSTM inputs
  for (int e = tid; e < MROWS * HDIM; e += 256) {
    const int row = e >> 8, hh = e & 255;
    int s = seqbase + row; if (s > N - 1) s = N - 1;
    ihid[(size_t)s * HDIM + hh] = hbuf[row][hh];
  }
}

// ----------------------------------------------------------- instruction LSTM
__global__ __launch_bounds__(256)
void instr_lstm_kernel(const int* __restrict__ ins_len,        // [B]
                       const int* __restrict__ bounds,         // [B+1]
                       const unsigned short* __restrict__ ihid,// [N,H] bf16
                       const unsigned short* __restrict__ wih, // [4H,H] bf16
                       const unsigned short* __restrict__ whh, // [4H,H] bf16
                       const float* __restrict__ bias,         // [4H]
                       const float* __restrict__ w_lin,        // [H]
                       const float* __restrict__ b_lin,        // [1]
                       float* __restrict__ out,                // [B]
                       int N, int B) {
  __shared__ unsigned short hbuf[MROWS][HPAD];
  __shared__ float cbuf[MROWS][CPAD];
  __shared__ int len_s[MROWS];
  __shared__ int bnd_s[MROWS];

  const int tid  = threadIdx.x;
  const int lane = tid & 31;
  const int wave = tid >> 5;
  const int l    = lane & 15;
  const int half = lane >> 4;
  const int seqbase = blockIdx.x * MROWS;

  float breg[8];
#pragma unroll
  for (int j = 0; j < 8; ++j) {
    const int gi = j >> 1, tc = j & 1;
    breg[j] = bias[gi * HDIM + wave * 32 + tc * 16 + l];
  }

  for (int e = tid; e < MROWS * HPAD; e += 256) ((unsigned short*)hbuf)[e] = 0;
  for (int e = tid; e < MROWS * CPAD; e += 256) ((float*)cbuf)[e] = 0.0f;
  if (tid < MROWS) {
    int s = seqbase + tid; if (s > B - 1) s = B - 1;
    len_s[tid] = ins_len[s];
    bnd_s[tid] = bounds[s];
  }
  __syncthreads();

  const unsigned short* wb_ih = wih + (size_t)(wave * 32 + l) * HDIM;
  const unsigned short* wb_hh = whh + (size_t)(wave * 32 + l) * HDIM;

  for (int m = 0; m < MMAX; ++m) {
    v8f acc[2][8];
#pragma unroll
    for (int mt = 0; mt < 2; ++mt)
#pragma unroll
      for (int j = 0; j < 8; ++j)
#pragma unroll
        for (int r = 0; r < 8; ++r) acc[mt][j][r] = 0.0f;

    // gather rows: instr_inputs[clip(bound + m, 0, N-1)]
    int g0 = bnd_s[l] + m;      if (g0 > N - 1) g0 = N - 1;
    int g1 = bnd_s[16 + l] + m; if (g1 > N - 1) g1 = N - 1;
    const unsigned short* xr0 = ihid + (size_t)g0 * HDIM;
    const unsigned short* xr1 = ihid + (size_t)g1 * HDIM;
    __builtin_prefetch(xr0, 0, 0);
    __builtin_prefetch(xr1, 0, 0);

    // -------- input path --------
#pragma unroll 1
    for (int kk = 0; kk < HDIM / 32; ++kk) {
      const int base1 = kk * 32 + half * 8;
      const int base2 = base1 + 16;
      const int wo    = kk * 32 + half * 8;
      FragBF a0, a1;
      a0.q[0] = *(const uint4*)(xr0 + base1);
      a0.q[1] = *(const uint4*)(xr0 + base2);
      a1.q[0] = *(const uint4*)(xr1 + base1);
      a1.q[1] = *(const uint4*)(xr1 + base2);
#pragma unroll
      for (int j = 0; j < 8; ++j) {
        const int gi = j >> 1, tc = j & 1;
        const unsigned short* wr = wb_ih + (size_t)(gi * 256 + tc * 16) * HDIM + wo;
        FragBF bf;
        bf.q[0] = *(const uint4*)wr;
        bf.q[1] = *(const uint4*)(wr + 16);
        acc[0][j] = __builtin_amdgcn_wmma_f32_16x16x32_bf16(
            false, a0.v, false, bf.v, (short)0, acc[0][j], false, false);
        acc[1][j] = __builtin_amdgcn_wmma_f32_16x16x32_bf16(
            false, a1.v, false, bf.v, (short)0, acc[1][j], false, false);
      }
    }

    // -------- recurrent path --------
#pragma unroll 1
    for (int kk = 0; kk < HDIM / 32; ++kk) {
      const int base1 = kk * 32 + half * 8;
      const int base2 = base1 + 16;
      const int wo    = kk * 32 + half * 8;
      FragBF a0, a1;
      a0.q[0] = *(const uint4*)&hbuf[l][base1];
      a0.q[1] = *(const uint4*)&hbuf[l][base2];
      a1.q[0] = *(const uint4*)&hbuf[16 + l][base1];
      a1.q[1] = *(const uint4*)&hbuf[16 + l][base2];
#pragma unroll
      for (int j = 0; j < 8; ++j) {
        const int gi = j >> 1, tc = j & 1;
        const unsigned short* wr = wb_hh + (size_t)(gi * 256 + tc * 16) * HDIM + wo;
        FragBF bf;
        bf.q[0] = *(const uint4*)wr;
        bf.q[1] = *(const uint4*)(wr + 16);
        acc[0][j] = __builtin_amdgcn_wmma_f32_16x16x32_bf16(
            false, a0.v, false, bf.v, (short)0, acc[0][j], false, false);
        acc[1][j] = __builtin_amdgcn_wmma_f32_16x16x32_bf16(
            false, a1.v, false, bf.v, (short)0, acc[1][j], false, false);
      }
    }

    __syncthreads();

#pragma unroll
    for (int mt = 0; mt < 2; ++mt) {
#pragma unroll
      for (int tc = 0; tc < 2; ++tc) {
        const int hh = wave * 32 + tc * 16 + l;
#pragma unroll
        for (int r = 0; r < 8; ++r) {
          const int row = mt * 16 + r + 8 * half;
          const float ig = acc[mt][0 + tc][r] + breg[0 + tc];
          const float fg = acc[mt][2 + tc][r] + breg[2 + tc];
          const float gg = acc[mt][4 + tc][r] + breg[4 + tc];
          const float og = acc[mt][6 + tc][r] + breg[6 + tc];
          const float co = cbuf[row][hh];
          const float cn = sigmoidf_(fg) * co + sigmoidf_(ig) * tanhf(gg);
          const float hn = sigmoidf_(og) * tanhf(cn);
          if (m < len_s[row]) {
            cbuf[row][hh] = cn;
            hbuf[row][hh] = f2bf(hn);
          }
        }
      }
    }
    __syncthreads();
  }

  // final linear head: out = h_final @ w_lin^T + b_lin
  if (tid < MROWS && seqbase + tid < B) {
    float s = b_lin[0];
    for (int k = 0; k < HDIM; ++k) s += bf2f(hbuf[tid][k]) * w_lin[k];
    out[seqbase + tid] = s;
  }
}

// ------------------------------------------------------------------- launcher
extern "C" void kernel_launch(void* const* d_in, const int* in_sizes, int n_in,
                              void* d_out, int out_size, void* d_ws, size_t ws_size,
                              hipStream_t stream) {
  (void)n_in; (void)out_size; (void)ws_size;

  const int* tokens    = (const int*)d_in[0];
  const int* tok_len   = (const int*)d_in[1];
  const int* ins_len   = (const int*)d_in[2];
  const int* bounds    = (const int*)d_in[3];
  const float* emb     = (const float*)d_in[4];
  const float* w_ih_tok= (const float*)d_in[5];
  const float* w_hh_tok= (const float*)d_in[6];
  const float* b_ih_tok= (const float*)d_in[7];
  const float* b_hh_tok= (const float*)d_in[8];
  const float* w_ih_ins= (const float*)d_in[9];
  const float* w_hh_ins= (const float*)d_in[10];
  const float* b_ih_ins= (const float*)d_in[11];
  const float* b_hh_ins= (const float*)d_in[12];
  const float* w_lin   = (const float*)d_in[13];
  const float* b_lin   = (const float*)d_in[14];
  float* out = (float*)d_out;

  const int N = in_sizes[0] / TMAX;     // number of instructions (16512)
  const int B = in_sizes[2];            // number of basic blocks (256)
  const int WNUM = GDIM * EDIM;         // elems per weight matrix (262144)
  const int ENUM = VDIM * EDIM;         // embedding elems (524288)

  // workspace carve-out
  char* ws = (char*)d_ws;
  unsigned short* wih_tok_bf = (unsigned short*)ws; ws += (size_t)WNUM * 2;
  unsigned short* whh_tok_bf = (unsigned short*)ws; ws += (size_t)WNUM * 2;
  unsigned short* wih_ins_bf = (unsigned short*)ws; ws += (size_t)WNUM * 2;
  unsigned short* whh_ins_bf = (unsigned short*)ws; ws += (size_t)WNUM * 2;
  unsigned short* emb_bf     = (unsigned short*)ws; ws += (size_t)ENUM * 2;
  float* bias_tok = (float*)ws; ws += (size_t)GDIM * 4;
  float* bias_ins = (float*)ws; ws += (size_t)GDIM * 4;
  unsigned short* ihid = (unsigned short*)ws;   // [N, H] bf16

  const int CT = 256;
  cvt_f32_bf16<<<(WNUM + CT - 1) / CT, CT, 0, stream>>>(w_ih_tok, wih_tok_bf, WNUM);
  cvt_f32_bf16<<<(WNUM + CT - 1) / CT, CT, 0, stream>>>(w_hh_tok, whh_tok_bf, WNUM);
  cvt_f32_bf16<<<(WNUM + CT - 1) / CT, CT, 0, stream>>>(w_ih_ins, wih_ins_bf, WNUM);
  cvt_f32_bf16<<<(WNUM + CT - 1) / CT, CT, 0, stream>>>(w_hh_ins, whh_ins_bf, WNUM);
  cvt_f32_bf16<<<(ENUM + CT - 1) / CT, CT, 0, stream>>>(emb, emb_bf, ENUM);
  add_bias<<<(GDIM + CT - 1) / CT, CT, 0, stream>>>(b_ih_tok, b_hh_tok, bias_tok, GDIM);
  add_bias<<<(GDIM + CT - 1) / CT, CT, 0, stream>>>(b_ih_ins, b_hh_ins, bias_ins, GDIM);

  token_lstm_kernel<<<(N + MROWS - 1) / MROWS, 256, 0, stream>>>(
      tokens, tok_len, emb_bf, wih_tok_bf, whh_tok_bf, bias_tok, ihid, N);

  instr_lstm_kernel<<<(B + MROWS - 1) / MROWS, 256, 0, stream>>>(
      ins_len, bounds, ihid, wih_ins_bf, whh_ins_bf, bias_ins,
      w_lin, b_lin, out, N, B);
}
